// Convpass_swin_hypernet_fuse_22359599743389
// MI455X (gfx1250) — matrix operand order repaired and checked by reading, not compiled, not run on hardware
//
#include <hip/hip_runtime.h>
#include <hip/hip_bf16.h>

// ---------------------------------------------------------------------------
// Convpass + swin hypernet fused pipeline for gfx1250 (MI455X).
//   B=32, H=W=28, C=768, D=EMB=64, P = B*H*W = 25088.
//
// Big GEMMs use V_WMMA_F32_16X16X32_BF16 with hi/lo split operands
// ("bf16x3": A*B ~= Ahi*Bhi + Alo*Bhi + Ahi*Blo, fp32 accumulate, ~1e-5 rel
// error).  Every hi/lo tensor is ONE buffer with the lo plane at a constant
// element offset (< 24-bit byte offset), so hi+lo loads share one address
// VGPR pair and lo folds into the instruction immediate -> no spills.
//
// K0 k_wprep   : pre-split static weights ([meta_w1;down_w] -> wcat, up_w).
// K1 k_proj128 : Y[25088x128] = X @ wcat^T, 32x64 C block/wave, A split
//                in-loop.  nt=0 -> relu -> per-sample sums; nt=1 -> qgelu ->
//                xact hi/lo.
// K2 k_fuse    : fused[32x64] = layer_emb + meta_b2 + (hsum/784) @ meta_w2^T
// K3 k_hyper   : convw = fused @ hyper_w^T + hyper_b (fp32 16x16x4 WMMA),
//                stores hi/lo bf16 permuted to k=(tap,din).
// K4 k_conv    : per-sample implicit-GEMM 3x3 conv (M=64,K=576,N=784),
//                operands pre-split; tap-hoisted masks; qgelu -> yact hi/lo.
// K5 k_up      : out = yact @ up_w^T + up_b, 32x64 C block/wave.
// ---------------------------------------------------------------------------

typedef __attribute__((ext_vector_type(2)))  float  v2f;
typedef __attribute__((ext_vector_type(8)))  float  v8f;
typedef __attribute__((ext_vector_type(16))) __bf16 v16bf;

#define NPOS   25088
#define CIN    768
#define DDIM   64
#define HWDIM  28
#define SPP    784

// lo-plane element offsets inside each combined hi/lo buffer
#define XACT_LO  1605632u   // 25088*64   (byte off 3211264  < 2^23)
#define YACT_LO  1605632u
#define CONVW_LO 1179648u   // 32*36864   (byte off 2359296  < 2^23)
#define WCAT_LO  98304u     // 128*768    (byte off 196608)
#define UP_LO    49152u     // 768*64     (byte off 98304)

__device__ __forceinline__ float qgelu(float x) {
  const float e = __expf(-1.702f * x);
  return x * __builtin_amdgcn_rcpf(1.0f + e);
}

__device__ __forceinline__ v8f wmma4(v2f a, v2f b, v8f c) {
  return __builtin_amdgcn_wmma_f32_16x16x4_f32(false, a, false, b, (short)0, c,
                                               false, false);
}
__device__ __forceinline__ v8f wmma32(v16bf a, v16bf b, v8f c) {
  return __builtin_amdgcn_wmma_f32_16x16x32_bf16(false, a, false, b, (short)0,
                                                 c, false, false);
}

__device__ __forceinline__ v2f ld2(const float* p) {
  const float2 v = *(const float2*)p;
  v2f r = {v.x, v.y};
  return r;
}

// split fp32 into bf16 hi + bf16 residual lo
__device__ __forceinline__ void split1(float f, __bf16& h, __bf16& l) {
  h = (__bf16)f;
  l = (__bf16)(f - (float)h);
}

// A operand (16-bit 16x32 layout): elements 0..7 = K[k0+8h .. +7],
// elements 8..15 = K[k0+16+8h .. +7]  (h = lane>=16)
__device__ __forceinline__ v16bf loadA_bf(const __bf16* row, int k0, int half) {
  union { uint4 u[2]; v16bf v; } c;
  c.u[0] = *(const uint4*)(row + k0 + 8 * half);
  c.u[1] = *(const uint4*)(row + k0 + 16 + 8 * half);
  return c.v;
}
// B operand (16-bit 32x16 layout): elements 0..7 = K[k0+8h .. +7],
// elements 8..15 = K[k0+16+8h .. +7]  (same order as A)
__device__ __forceinline__ v16bf loadB_bf(const __bf16* col, int k0, int half) {
  union { uint4 u[2]; v16bf v; } c;
  c.u[0] = *(const uint4*)(col + k0 + 8 * half);
  c.u[1] = *(const uint4*)(col + k0 + 16 + 8 * half);
  return c.v;
}
// load 16 fp32 in A-chunk order and split into hi/lo v16bf
__device__ __forceinline__ void loadA_split(const float* row, int k0, int half,
                                            v16bf& hi, v16bf& lo) {
  const float* p0 = row + k0 + 8 * half;
  const float* p1 = row + k0 + 16 + 8 * half;
  const float4 f0 = *(const float4*)(p0);
  const float4 f1 = *(const float4*)(p0 + 4);
  const float4 f2 = *(const float4*)(p1);
  const float4 f3 = *(const float4*)(p1 + 4);
  const float buf[16] = {f0.x, f0.y, f0.z, f0.w, f1.x, f1.y, f1.z, f1.w,
                         f2.x, f2.y, f2.z, f2.w, f3.x, f3.y, f3.z, f3.w};
#pragma unroll
  for (int e = 0; e < 16; ++e) {
    __bf16 h, l;
    split1(buf[e], h, l);
    hi[e] = h;
    lo[e] = l;
  }
}

// ---------------------------------------------------------------------------
// K0: pre-split static weights.  147456 elements, 576 blocks * 256.
// ---------------------------------------------------------------------------
__global__ void k_wprep(const float* __restrict__ meta_w1,
                        const float* __restrict__ down_w,
                        const float* __restrict__ up_w,
                        __bf16* __restrict__ wcat, __bf16* __restrict__ upw) {
  const int idx = blockIdx.x * blockDim.x + threadIdx.x;
  if (idx < 128 * CIN) {
    const float v = (idx < 64 * CIN) ? meta_w1[idx] : down_w[idx - 64 * CIN];
    split1(v, wcat[idx], wcat[idx + WCAT_LO]);
  } else if (idx < 128 * CIN + CIN * DDIM) {
    const int j = idx - 128 * CIN;
    split1(up_w[j], upw[j], upw[j + UP_LO]);
  }
}

// ---------------------------------------------------------------------------
// K1: 784 M-tiles(32) x 2 N-tiles(64) = 1568 waves = 196 blocks * 8 waves.
// ---------------------------------------------------------------------------
__global__ void k_proj128(const float* __restrict__ x,
                          const __bf16* __restrict__ wcat,
                          const float* __restrict__ meta_b1,
                          const float* __restrict__ down_b,
                          __bf16* __restrict__ xact,
                          float* __restrict__ hsum) {
  const int lane = threadIdx.x & 31;
  const int wid  = blockIdx.x * (blockDim.x >> 5) + (threadIdx.x >> 5);
  const int mt   = wid >> 1;          // 0..783
  const int nt   = wid & 1;           // 0: meta, 1: adapter-down
  const int half = lane >> 4, lr = lane & 15;
  const int row0 = mt * 32;

  const float* a0 = x + (size_t)(row0 + lr) * CIN;
  const float* a1 = x + (size_t)(row0 + 16 + lr) * CIN;
  // one address for all 8 weight tiles: j/lo planes are constant offsets
  const __bf16* wb = wcat + (size_t)(nt * 64 + lr) * CIN;

  v8f acc[2][4];
#pragma unroll
  for (int i = 0; i < 2; ++i)
#pragma unroll
    for (int j = 0; j < 4; ++j) acc[i][j] = {};

#pragma unroll 1
  for (int k0 = 0; k0 < CIN; k0 += 32) {
    v16bf A0h, A0l, A1h, A1l;
    loadA_split(a0, k0, half, A0h, A0l);
    loadA_split(a1, k0, half, A1h, A1l);
#pragma unroll
    for (int j = 0; j < 4; ++j) {
      const v16bf Bh = loadB_bf(wb + j * 16 * CIN, k0, half);
      const v16bf Bl = loadB_bf(wb + j * 16 * CIN + WCAT_LO, k0, half);
      acc[0][j] = wmma32(A0h, Bh, acc[0][j]);
      acc[0][j] = wmma32(A0l, Bh, acc[0][j]);
      acc[0][j] = wmma32(A0h, Bl, acc[0][j]);
      acc[1][j] = wmma32(A1h, Bh, acc[1][j]);
      acc[1][j] = wmma32(A1l, Bh, acc[1][j]);
      acc[1][j] = wmma32(A1h, Bl, acc[1][j]);
    }
  }

  if (nt == 0) {
    const int blo = row0 / SPP;           // 784 % 16 == 0: 16-row halves are
    const int bhi = (row0 + 16) / SPP;    // sample-pure
#pragma unroll
    for (int j = 0; j < 4; ++j) {
      const int n = j * 16 + lr;
      const float bj = meta_b1[n];
      float s0 = 0.f, s1 = 0.f;
#pragma unroll
      for (int r = 0; r < 8; ++r) {
        s0 += fmaxf(acc[0][j][r] + bj, 0.f);
        s1 += fmaxf(acc[1][j][r] + bj, 0.f);
      }
      s0 += __shfl_xor(s0, 16, 32);
      s1 += __shfl_xor(s1, 16, 32);
      if (half == 0) {
        atomicAdd(&hsum[blo * DDIM + n], s0);
        atomicAdd(&hsum[bhi * DDIM + n], s1);
      }
    }
  } else {
#pragma unroll
    for (int j = 0; j < 4; ++j) {
      const int d = j * 16 + lr;
      const float bj = down_b[d];
#pragma unroll
      for (int r = 0; r < 8; ++r) {
        const size_t p0 = (size_t)(row0 + r + 8 * half) * DDIM + d;
        const size_t p1 = (size_t)(row0 + 16 + r + 8 * half) * DDIM + d;
        split1(qgelu(acc[0][j][r] + bj), xact[p0], xact[p0 + XACT_LO]);
        split1(qgelu(acc[1][j][r] + bj), xact[p1], xact[p1 + XACT_LO]);
      }
    }
  }
}

// ---------------------------------------------------------------------------
// K2: fused = layer_emb + meta_b2 + (hsum/784) @ meta_w2^T
// ---------------------------------------------------------------------------
__global__ void k_fuse(const float* __restrict__ hsum,
                       const float* __restrict__ meta_w2,
                       const float* __restrict__ meta_b2,
                       const float* __restrict__ layer_emb,
                       float* __restrict__ fused) {
  const int idx = blockIdx.x * blockDim.x + threadIdx.x;
  if (idx >= 32 * DDIM) return;
  const int b = idx >> 6, p = idx & 63;
  const float inv = 1.0f / (float)SPP;
  float acc = 0.0f;
#pragma unroll 8
  for (int o = 0; o < DDIM; ++o)
    acc += (hsum[b * DDIM + o] * inv) * meta_w2[p * DDIM + o];
  fused[idx] = acc + meta_b2[p] + layer_emb[p];
}

// ---------------------------------------------------------------------------
// K3: convw[32 x 36864] = fused @ hyper_w^T + hyper_b, fp32 16x16x4 WMMA,
//     32x16 block per wave (hyper_w streamed once).  288 blocks * 8 waves.
//     Stores bf16 hi/lo, permuted (dout,din,kh,kw) -> dout*576+(kh*3+kw)*64+din
// ---------------------------------------------------------------------------
__global__ void k_hyper(const float* __restrict__ fused,
                        const float* __restrict__ hyper_w,
                        const float* __restrict__ hyper_b,
                        __bf16* __restrict__ convw) {
  const int lane = threadIdx.x & 31;
  const int nt   = blockIdx.x * (blockDim.x >> 5) + (threadIdx.x >> 5); // 0..2303
  const int half = lane >> 4, lr = lane & 15;
  const int col0 = nt * 16;

  const float* a0 = fused + (size_t)lr * DDIM;          // samples 0..15
  const float* a1 = fused + (size_t)(16 + lr) * DDIM;   // samples 16..31
  const float* wr = hyper_w + (size_t)(col0 + lr) * DDIM;

  v8f c0 = {}, c1 = {};
#pragma unroll
  for (int k0 = 0; k0 < DDIM; k0 += 4) {
    const int ko = k0 + 2 * half;
    const v2f A0 = ld2(a0 + ko), A1 = ld2(a1 + ko);
    const v2f B  = ld2(wr + ko);
    c0 = wmma4(A0, B, c0);
    c1 = wmma4(A1, B, c1);
  }

  const int j    = col0 + lr;          // flat (dout,din,kh,kw)
  const int dout = j / 576;
  const int rem  = j - dout * 576;
  const int din  = rem / 9;
  const int tap  = rem - din * 9;
  const size_t jp = (size_t)dout * 576 + tap * 64 + din; // k=(tap,din) order
  const float bias = hyper_b[j];
#pragma unroll
  for (int r = 0; r < 8; ++r) {
    const size_t q0 = (size_t)(r + 8 * half) * 36864 + jp;
    const size_t q1 = (size_t)(16 + r + 8 * half) * 36864 + jp;
    split1(c0[r] + bias, convw[q0], convw[q0 + CONVW_LO]);
    split1(c1[r] + bias, convw[q1], convw[q1 + CONVW_LO]);
  }
}

// ---------------------------------------------------------------------------
// K4: per-sample 3x3 conv as implicit GEMM (k = tap*64 + din), bf16x3.
//     32 samples * 2 M-tiles(32) * 49 N-tiles = 3136 waves = 392 blocks.
// ---------------------------------------------------------------------------
__global__ void k_conv(const __bf16* __restrict__ convw,
                       const __bf16* __restrict__ xact,
                       __bf16* __restrict__ yact) {
  const int lane = threadIdx.x & 31;
  const int wid  = blockIdx.x * (blockDim.x >> 5) + (threadIdx.x >> 5);
  const int b    = wid / 98;
  const int rem  = wid % 98;
  const int mt   = rem / 49;          // 0..1 (32 d_out rows each)
  const int nt   = rem % 49;
  const int half = lane >> 4, lr = lane & 15;

  const int s  = nt * 16 + lr;
  const int ho = s / HWDIM, wo = s % HWDIM;

  const __bf16* a0 = convw + (size_t)b * 36864 + (size_t)(mt * 32 + lr) * 576;
  const __bf16* a1 = a0 + 16 * 576;
  const __bf16* xb = xact + (size_t)b * SPP * DDIM;

  v8f c0 = {}, c1 = {};
  const v16bf vz = {};
#pragma unroll 1
  for (int tap = 0; tap < 9; ++tap) {
    const int kh = tap / 3, kw = tap - 3 * (tap / 3);
    const int hi = ho + kh - 1, wi = wo + kw - 1;
    const bool valid = ((unsigned)hi < HWDIM) & ((unsigned)wi < HWDIM);
    const int  off   = valid ? ((hi * HWDIM + wi) << 6) : 0;   // safe addr
#pragma unroll
    for (int dc = 0; dc < 2; ++dc) {
      const int k0 = tap * 64 + dc * 32;   // A element index
      const int bk = dc * 32;              // xact index within row
      const v16bf A0h = loadA_bf(a0, k0, half);
      const v16bf A0l = loadA_bf(a0 + CONVW_LO, k0, half);
      const v16bf A1h = loadA_bf(a1, k0, half);
      const v16bf A1l = loadA_bf(a1 + CONVW_LO, k0, half);
      v16bf Bh = loadB_bf(xb + off, bk, half);
      v16bf Bl = loadB_bf(xb + off + XACT_LO, bk, half);
      Bh = valid ? Bh : vz;
      Bl = valid ? Bl : vz;
      c0 = wmma32(A0h, Bh, c0);
      c0 = wmma32(A0l, Bh, c0);
      c0 = wmma32(A0h, Bl, c0);
      c1 = wmma32(A1h, Bh, c1);
      c1 = wmma32(A1l, Bh, c1);
      c1 = wmma32(A1h, Bl, c1);
    }
  }
#pragma unroll
  for (int r = 0; r < 8; ++r) {
    const size_t base = ((size_t)b * SPP + s) * DDIM;
    const int d0 = mt * 32 + r + 8 * half;
    const int d1 = d0 + 16;
    split1(qgelu(c0[r]), yact[base + d0], yact[base + d0 + YACT_LO]);
    split1(qgelu(c1[r]), yact[base + d1], yact[base + d1 + YACT_LO]);
  }
}

// ---------------------------------------------------------------------------
// K5: out = yact @ up_w^T + up_b, bf16x3, 32x64 C block per wave.
//     784 * 12 = 9408 waves = 1176 blocks * 8 waves.
// ---------------------------------------------------------------------------
__global__ void k_up(const __bf16* __restrict__ yact,
                     const __bf16* __restrict__ upw,
                     const float* __restrict__ up_b,
                     float* __restrict__ out) {
  const int lane = threadIdx.x & 31;
  const int wid  = blockIdx.x * (blockDim.x >> 5) + (threadIdx.x >> 5);
  const int mt   = wid / 12;
  const int nt   = wid % 12;
  const int half = lane >> 4, lr = lane & 15;
  const int row0 = mt * 32, col0 = nt * 64;

  const __bf16* a0 = yact + (size_t)(row0 + lr) * DDIM;
  const __bf16* a1 = yact + (size_t)(row0 + 16 + lr) * DDIM;
  const __bf16* wb = upw + (size_t)(col0 + lr) * DDIM;

  v8f acc[2][4];
#pragma unroll
  for (int i = 0; i < 2; ++i)
#pragma unroll
    for (int j = 0; j < 4; ++j) acc[i][j] = {};

#pragma unroll
  for (int k0 = 0; k0 < DDIM; k0 += 32) {
    const v16bf A0h = loadA_bf(a0, k0, half);
    const v16bf A0l = loadA_bf(a0 + YACT_LO, k0, half);
    const v16bf A1h = loadA_bf(a1, k0, half);
    const v16bf A1l = loadA_bf(a1 + YACT_LO, k0, half);
#pragma unroll
    for (int j = 0; j < 4; ++j) {
      const v16bf Bh = loadB_bf(wb + j * 16 * DDIM, k0, half);
      const v16bf Bl = loadB_bf(wb + j * 16 * DDIM + UP_LO, k0, half);
      acc[0][j] = wmma32(A0h, Bh, acc[0][j]);
      acc[0][j] = wmma32(A0l, Bh, acc[0][j]);
      acc[0][j] = wmma32(A0h, Bl, acc[0][j]);
      acc[1][j] = wmma32(A1h, Bh, acc[1][j]);
      acc[1][j] = wmma32(A1l, Bh, acc[1][j]);
      acc[1][j] = wmma32(A1h, Bl, acc[1][j]);
    }
  }

#pragma unroll
  for (int j = 0; j < 4; ++j) {
    const int n = col0 + j * 16 + lr;
    const float bj = up_b[n];
#pragma unroll
    for (int r = 0; r < 8; ++r) {
      const size_t p0 = (size_t)(row0 + r + 8 * half) * CIN;
      const size_t p1 = (size_t)(row0 + 16 + r + 8 * half) * CIN;
      out[p0 + n] = acc[0][j][r] + bj;
      out[p1 + n] = acc[1][j][r] + bj;
    }
  }
}

// ---------------------------------------------------------------------------
extern "C" void kernel_launch(void* const* d_in, const int* in_sizes, int n_in,
                              void* d_out, int out_size, void* d_ws, size_t ws_size,
                              hipStream_t stream) {
  const float* x         = (const float*)d_in[0];
  const float* meta_w1   = (const float*)d_in[1];
  const float* meta_b1   = (const float*)d_in[2];
  const float* meta_w2   = (const float*)d_in[3];
  const float* meta_b2   = (const float*)d_in[4];
  const float* layer_emb = (const float*)d_in[5];
  const float* hyper_w   = (const float*)d_in[6];
  const float* hyper_b   = (const float*)d_in[7];
  const float* down_w    = (const float*)d_in[8];
  const float* down_b    = (const float*)d_in[9];
  const float* up_w      = (const float*)d_in[10];
  const float* up_b      = (const float*)d_in[11];
  float* out = (float*)d_out;

  // workspace layout (bytes), all offsets 256B-aligned, ~18.2 MB total.
  // each bf16 tensor holds its hi plane at +0 and lo plane at +N elements.
  char* ws = (char*)d_ws;
  __bf16* xact  = (__bf16*)(ws + 0);         // 2*1605632*2 = 6422528
  __bf16* yact  = (__bf16*)(ws + 6422528);   // 6422528
  __bf16* convw = (__bf16*)(ws + 12845056);  // 2*1179648*2 = 4718592
  __bf16* wcat  = (__bf16*)(ws + 17563648);  // 2*98304*2  = 393216
  __bf16* upw   = (__bf16*)(ws + 17956864);  // 2*49152*2  = 196608
  float*  hsum  = (float*) (ws + 18153472);  // 2048*4
  float*  fused = (float*) (ws + 18161664);  // 2048*4

  hipMemsetAsync(hsum, 0, 32 * DDIM * sizeof(float), stream);  // capture-safe

  k_wprep<<<576, 256, 0, stream>>>(meta_w1, down_w, up_w, wcat, upw);
  k_proj128<<<196, 256, 0, stream>>>(x, wcat, meta_b1, down_b, xact, hsum);
  k_fuse<<<8, 256, 0, stream>>>(hsum, meta_w2, meta_b2, layer_emb, fused);
  k_hyper<<<288, 256, 0, stream>>>(fused, hyper_w, hyper_b, convw);
  k_conv<<<392, 256, 0, stream>>>(convw, xact, yact);
  k_up<<<1176, 256, 0, stream>>>(yact, upw, up_b, out);
}
